// ParticleNet_70927089926266
// MI455X (gfx1250) — matrix operand order repaired and checked by reading, not compiled, not run on hardware
//
#include <hip/hip_runtime.h>
#include <hip/hip_bf16.h>

// ---------------------------------------------------------------------------
// Types for CDNA5 WMMA (wave32, 16x16x32 bf16 -> f32)
// ---------------------------------------------------------------------------
typedef __attribute__((ext_vector_type(16))) __bf16 bf16x16;
typedef __attribute__((ext_vector_type(8)))  __bf16 bf16x8;
typedef __attribute__((ext_vector_type(8)))  float  f32x8;

#define BB   256
#define NN   128
#define KK   16

// ---------------------------------------------------------------------------
// WMMA fragment loaders (ISA 7.12.2 layouts)
// A (16x32 bf16, row-major source with leading dim `ld`):
//   lane<16: M=lane,     K = k0+{0..7, 16..23}
//   lane>=16:M=lane-16,  K = k0+{8..15, 24..31}
// B (32x16 bf16 = W^T slice; W stored row-major (Co,Ci) so column N of B is
//   a contiguous row of W):
//   lane<16: N=lane,     K = k0+{0..7, 16..23}
//   lane>=16:N=lane-16,  K = k0+{8..15, 24..31}
// ---------------------------------------------------------------------------
__device__ __forceinline__ bf16x16 load_fragA(const __bf16* base, int ld, int row0, int k0) {
    int lane = threadIdx.x & 31;
    const __bf16* p = base + (size_t)(row0 + (lane & 15)) * ld + k0 + ((lane >> 4) << 3);
    bf16x8 lo = *(const bf16x8*)(p);
    bf16x8 hi = *(const bf16x8*)(p + 16);
    bf16x16 r;
#pragma unroll
    for (int i = 0; i < 8; ++i) { r[i] = lo[i]; r[i + 8] = hi[i]; }
    return r;
}

__device__ __forceinline__ bf16x16 load_fragB(const __bf16* W, int ld, int n0, int k0) {
    int lane = threadIdx.x & 31;
    const __bf16* p = W + (size_t)(n0 + (lane & 15)) * ld + k0 + ((lane >> 4) << 3);
    bf16x8 lo = *(const bf16x8*)(p);
    bf16x8 hi = *(const bf16x8*)(p + 16);
    bf16x16 r;
#pragma unroll
    for (int i = 0; i < 8; ++i) { r[i] = lo[i]; r[i + 8] = hi[i]; }
    return r;
}

// ---------------------------------------------------------------------------
// Register-blocked fused GEMM + BN + ReLU layer (128 rows, 8 waves):
//   Y(128 x CO, LDS bf16) = relu( X(128 x CI, LDS bf16) * W^T * scale + bias )
// Column tiles processed in chunks of <=8 so the A fragment is loaded from
// LDS once per k-step and reused by 8 independent WMMAs (64 VGPR acc block).
// ---------------------------------------------------------------------------
template <int CI, int CO>
__device__ __forceinline__ void gemm_bn_relu_128_t(const __bf16* Xl, __bf16* Yl,
                                                   const __bf16* __restrict__ W,
                                                   const float* __restrict__ scale,
                                                   const float* __restrict__ bias) {
    constexpr int NT    = CO / 16;
    constexpr int CHUNK = (NT < 8) ? NT : 8;
    int wave = threadIdx.x >> 5;
    int lane = threadIdx.x & 31;
    int row0 = wave * 16;
    int mbase = row0 + ((lane >> 4) << 3);

#pragma unroll 1
    for (int nc = 0; nc < NT; nc += CHUNK) {
        f32x8 acc[CHUNK];
#pragma unroll
        for (int j = 0; j < CHUNK; ++j) acc[j] = (f32x8){};

#pragma unroll 1
        for (int k0 = 0; k0 < CI; k0 += 32) {
            bf16x16 a = load_fragA(Xl, CI, row0, k0);
#pragma unroll
            for (int j = 0; j < CHUNK; ++j) {
                bf16x16 b = load_fragB(W, CI, (nc + j) * 16, k0);
                acc[j] = __builtin_amdgcn_wmma_f32_16x16x32_bf16(
                    false, a, false, b, (short)0, acc[j], false, false);
            }
        }

#pragma unroll
        for (int j = 0; j < CHUNK; ++j) {
            int ch = (nc + j) * 16 + (lane & 15);
            float s  = scale[ch];
            float bi = bias[ch];
#pragma unroll
            for (int r = 0; r < 8; ++r) {
                float y = fmaf(acc[j][r], s, bi);
                y = y > 0.f ? y : 0.f;
                Yl[(size_t)(mbase + r) * CO + ch] = (__bf16)y;
            }
        }
    }
}

// ---------------------------------------------------------------------------
// Kernel 1: preprocessing.  points*mask, BN on features, counts.
// grid = B, block = 128 (one thread per point)
// ---------------------------------------------------------------------------
__global__ void prep_kernel(const float* __restrict__ points,
                            const float* __restrict__ features,
                            const float* __restrict__ mask,
                            const float* __restrict__ bn_scale,
                            const float* __restrict__ bn_bias,
                            float* __restrict__ pts2,
                            float* __restrict__ fts0,
                            float* __restrict__ counts) {
    int b = blockIdx.x, n = threadIdx.x;
    float m = mask[b * NN + n];
    pts2[(size_t)(b * NN + n) * 2 + 0] = points[(b * 2 + 0) * NN + n] * m;
    pts2[(size_t)(b * NN + n) * 2 + 1] = points[(b * 2 + 1) * NN + n] * m;
#pragma unroll
    for (int c = 0; c < 7; ++c) {
        float f = features[(size_t)(b * 7 + c) * NN + n];
        fts0[(size_t)(b * NN + n) * 7 + c] = fmaf(f, bn_scale[c], bn_bias[c]) * m;
    }
    if (n == 0) {
        float s = 0.f;
        for (int i = 0; i < NN; ++i) s += mask[b * NN + i];
        counts[b] = fmaxf(s, 1.f);
    }
}

// ---------------------------------------------------------------------------
// Kernel 2: weight conversion f32 -> bf16, zero-padded along Ci to Ci_pad
// ---------------------------------------------------------------------------
__global__ void wconv_kernel(const float* __restrict__ W, __bf16* __restrict__ Wb,
                             int Co, int Ci, int Ci_pad) {
    int t = blockIdx.x * blockDim.x + threadIdx.x;
    int total = Co * Ci_pad;
    if (t >= total) return;
    int o = t / Ci_pad, c = t % Ci_pad;
    Wb[t] = (c < Ci) ? (__bf16)W[(size_t)o * Ci + c] : (__bf16)0.f;
}

// ---------------------------------------------------------------------------
// Kernel 3: kNN (top-16 by negative squared distance; masked points pushed
// out by +1e9 per coordinate, mirroring the reference's coord_shift).
// grid = B, block = 256, dyn LDS = (128*C + 128*128)*4
// ---------------------------------------------------------------------------
__global__ void __launch_bounds__(256, 1)
knn_kernel(const float* __restrict__ feat, const float* __restrict__ mask,
           int C, int* __restrict__ idx) {
    extern __shared__ char smem[];
    float* F = (float*)smem;            // 128 x C (shifted)
    float* D = F + (size_t)NN * C;      // 128 x 128
    int b = blockIdx.x, tid = threadIdx.x;

    for (int e = tid; e < NN * C; e += 256) {
        int n = e / C;
        float shift = (mask[b * NN + n] == 0.f) ? 1.0e9f : 0.f;
        F[e] = feat[(size_t)(b * NN + n) * C + e % C] + shift;
    }
    __syncthreads();

    for (int p = tid; p < NN * NN; p += 256) {
        int n = p >> 7, m = p & 127;
        float d;
        if (m == n) d = -3.4e38f;
        else {
            float s = 0.f;
            for (int c = 0; c < C; ++c) {
                float df = F[n * C + c] - F[m * C + c];
                s = fmaf(df, df, s);
            }
            d = -s;
        }
        D[p] = d;
    }
    __syncthreads();

    if (tid < NN) {
        float bd[KK]; int bi[KK];
#pragma unroll
        for (int i = 0; i < KK; ++i) { bd[i] = -3.4e38f; bi[i] = 0; }
        for (int m = 0; m < NN; ++m) {
            if (m == tid) continue;
            float d = D[tid * NN + m];
            if (d > bd[KK - 1]) {
                int j = KK - 1;
                while (j > 0 && bd[j - 1] < d) { bd[j] = bd[j - 1]; bi[j] = bi[j - 1]; --j; }
                bd[j] = d; bi[j] = m;
            }
        }
#pragma unroll
        for (int i = 0; i < KK; ++i) idx[(size_t)(b * NN + tid) * KK + i] = bi[i];
    }
}

// ---------------------------------------------------------------------------
// Kernel 4: fused EdgeConv block. One WG = 8 points x 16 neighbors = 128 rows.
// LDS: X0 (128 x CI1 bf16) | HA (128 x CO bf16) | HB (128 x CO bf16) | ctr f32
// grid = B * 16, block = 256 (8 waves)
// ---------------------------------------------------------------------------
template <int CIN, int CI1, int CO>
__global__ void __launch_bounds__(256, 1)
edgeconv_kernel(const float* __restrict__ fts, const int* __restrict__ idx,
                const float* __restrict__ mask,
                const __bf16* __restrict__ W1, const __bf16* __restrict__ W2,
                const __bf16* __restrict__ W3,
                const float* __restrict__ s1, const float* __restrict__ b1,
                const float* __restrict__ s2, const float* __restrict__ b2,
                const float* __restrict__ s3, const float* __restrict__ b3,
                const float* __restrict__ Wsc, const float* __restrict__ ssc,
                const float* __restrict__ bsc,
                float* __restrict__ out) {
    extern __shared__ char smem[];
    __bf16* X0 = (__bf16*)smem;
    __bf16* HA = X0 + (size_t)128 * CI1;
    __bf16* HB = HA + (size_t)128 * CO;
    float*  ctr = (float*)(HB + (size_t)128 * CO);

    int b  = blockIdx.x >> 4;
    int p0 = (blockIdx.x & 15) * 8;
    int tid = threadIdx.x;

    // center features for the 8 points
    for (int e = tid; e < 8 * CIN; e += 256)
        ctr[e] = fts[(size_t)(b * NN + p0 + e / CIN) * CIN + e % CIN];
    __syncthreads();

    // build X0 = [ctr | nb - ctr], zero padded to CI1
    for (int e = tid; e < 128 * CI1; e += 256) {
        int r = e / CI1, c = e % CI1;
        int p = r >> 4, k = r & 15;
        float v = 0.f;
        if (c < CIN) {
            v = ctr[p * CIN + c];
        } else if (c < 2 * CIN) {
            int cc  = c - CIN;
            int nbr = idx[(size_t)(b * NN + p0 + p) * KK + k];
            v = fts[(size_t)(b * NN + nbr) * CIN + cc] - ctr[p * CIN + cc];
        }
        X0[e] = (__bf16)v;
    }
    __syncthreads();

    gemm_bn_relu_128_t<CI1, CO>(X0, HA, W1, s1, b1); __syncthreads();
    gemm_bn_relu_128_t<CO,  CO>(HA, HB, W2, s2, b2); __syncthreads();
    gemm_bn_relu_128_t<CO,  CO>(HB, HA, W3, s3, b3); __syncthreads();

    // mean over K + shortcut + relu + mask
    for (int e = tid; e < 8 * CO; e += 256) {
        int p = e / CO, c = e % CO;
        float m = 0.f;
#pragma unroll 4
        for (int k = 0; k < KK; ++k) m += (float)HA[(size_t)(p * KK + k) * CO + c];
        m *= (1.f / 16.f);
        float sc = 0.f;
        for (int j = 0; j < CIN; ++j) sc = fmaf(ctr[p * CIN + j], Wsc[(size_t)c * CIN + j], sc);
        float y = fmaf(sc, ssc[c], bsc[c]) + m;
        y = (y > 0.f ? y : 0.f) * mask[b * NN + p0 + p];
        out[(size_t)(b * NN + p0 + p) * CO + c] = y;
    }
}

// ---------------------------------------------------------------------------
// Kernel 5: concat out0|out1|out2 -> bf16 (B*N, 448)
// ---------------------------------------------------------------------------
__global__ void concat_kernel(const float* __restrict__ o0, const float* __restrict__ o1,
                              const float* __restrict__ o2, __bf16* __restrict__ xcat) {
    size_t t = (size_t)blockIdx.x * blockDim.x + threadIdx.x;
    size_t total = (size_t)BB * NN * 448;
    if (t >= total) return;
    size_t r = t / 448;
    int    c = (int)(t % 448);
    float v;
    if (c < 64)       v = o0[r * 64 + c];
    else if (c < 192) v = o1[r * 128 + (c - 64)];
    else              v = o2[r * 256 + (c - 192)];
    xcat[t] = (__bf16)v;
}

// ---------------------------------------------------------------------------
// Kernel 6: generic 128-row-tile WMMA GEMM (register-blocked):
//   Y = act( X(M x CI bf16) * W^T * scale + bias ) * rowscale
// grid.x = M/128, block = 256, dyn LDS = 128*CI*2
// ---------------------------------------------------------------------------
template <int CI, int CO>
__global__ void __launch_bounds__(256, 1)
gemm_kernel(const __bf16* __restrict__ X,
            const __bf16* __restrict__ W,
            const float* __restrict__ scale, const float* __restrict__ bias,
            int relu, const float* __restrict__ rowscale,
            float* __restrict__ Yf) {
    extern __shared__ char smem[];
    __bf16* Xl = (__bf16*)smem;
    int row0g = blockIdx.x * 128;
    int tid = threadIdx.x;

    // stage 128 x CI tile into LDS (16B chunks)
    const bf16x8* src = (const bf16x8*)(X + (size_t)row0g * CI);
    bf16x8* dst = (bf16x8*)Xl;
    constexpr int N16 = (128 * CI) >> 3;
    for (int i = tid; i < N16; i += 256) dst[i] = src[i];
    __syncthreads();

    constexpr int NT    = CO / 16;
    constexpr int CHUNK = (NT < 8) ? NT : 8;
    int wave = tid >> 5, lane = tid & 31;
    int row0 = wave * 16;
    int mbase = row0 + ((lane >> 4) << 3);

#pragma unroll 1
    for (int nc = 0; nc < NT; nc += CHUNK) {
        f32x8 acc[CHUNK];
#pragma unroll
        for (int j = 0; j < CHUNK; ++j) acc[j] = (f32x8){};

#pragma unroll 1
        for (int k0 = 0; k0 < CI; k0 += 32) {
            bf16x16 a = load_fragA(Xl, CI, row0, k0);
#pragma unroll
            for (int j = 0; j < CHUNK; ++j) {
                bf16x16 bfr = load_fragB(W, CI, (nc + j) * 16, k0);
                acc[j] = __builtin_amdgcn_wmma_f32_16x16x32_bf16(
                    false, a, false, bfr, (short)0, acc[j], false, false);
            }
        }

#pragma unroll
        for (int j = 0; j < CHUNK; ++j) {
            int ch = (nc + j) * 16 + (lane & 15);
            float s  = scale ? scale[ch] : 1.f;
            float bi = bias  ? bias[ch]  : 0.f;
#pragma unroll
            for (int r = 0; r < 8; ++r) {
                int row = row0g + mbase + r;
                float y = fmaf(acc[j][r], s, bi);
                if (relu) y = y > 0.f ? y : 0.f;
                if (rowscale) y *= rowscale[row];
                Yf[(size_t)row * CO + ch] = y;
            }
        }
    }
}

// ---------------------------------------------------------------------------
// Kernel 7: masked mean pool over N, writes f32 + bf16
// grid = B, block = 256
// ---------------------------------------------------------------------------
__global__ void pool_kernel(const float* __restrict__ fused, const float* __restrict__ counts,
                            float* __restrict__ pooled, __bf16* __restrict__ pooledb) {
    int b = blockIdx.x;
    float inv = 1.f / counts[b];
    for (int c = threadIdx.x; c < 384; c += 256) {
        float s = 0.f;
        for (int n = 0; n < NN; ++n) s += fused[(size_t)(b * NN + n) * 384 + c];
        float p = s * inv;
        pooled[(size_t)b * 384 + c]  = p;
        pooledb[(size_t)b * 384 + c] = (__bf16)p;
    }
}

// ---------------------------------------------------------------------------
// Kernel 8: fc2 (tiny: 256x5, dot length 256)
// ---------------------------------------------------------------------------
__global__ void fc2_kernel(const float* __restrict__ h, const float* __restrict__ W,
                           const float* __restrict__ bias, float* __restrict__ out) {
    int t = blockIdx.x * blockDim.x + threadIdx.x;
    if (t >= BB * 5) return;
    int b = t / 5, o = t % 5;
    float s = bias[o];
    for (int c = 0; c < 256; ++c) s = fmaf(h[(size_t)b * 256 + c], W[(size_t)o * 256 + c], s);
    out[t] = s;
}

// ---------------------------------------------------------------------------
// Host-side launch
// ---------------------------------------------------------------------------
extern "C" void kernel_launch(void* const* d_in, const int* in_sizes, int n_in,
                              void* d_out, int out_size, void* d_ws, size_t ws_size,
                              hipStream_t stream) {
    (void)in_sizes; (void)n_in; (void)out_size; (void)ws_size;

    // ---- input pointers (jax pytree flatten order: sorted dict keys) ----
    const float* points   = (const float*)d_in[0];
    const float* features = (const float*)d_in[1];
    const float* mask     = (const float*)d_in[2];
    // blocks[0].convs: (W,bias,scale) x3 -> 3..11 ; sc: 12,13,14
    // blocks[1]: 15..23 ; sc: 24,25,26
    // blocks[2]: 27..35 ; sc: 36,37,38
    // bn_fts: bias=39 scale=40 ; fc1: W=41 b=42 ; fc2: W=43 b=44
    // fusion: W=45 bias=46 scale=47
    const float* bn_bias  = (const float*)d_in[39];
    const float* bn_scale = (const float*)d_in[40];

    // ---- workspace layout ----
    char* ws = (char*)d_ws;
    size_t off = 0;
    auto alloc = [&](size_t bytes) -> char* {
        char* p = ws + off;
        off = (off + bytes + 255) & ~(size_t)255;
        return p;
    };
    float*  pts2   = (float*) alloc((size_t)BB * NN * 2 * 4);
    float*  fts0   = (float*) alloc((size_t)BB * NN * 7 * 4);
    int*    idx    = (int*)   alloc((size_t)BB * NN * KK * 4);
    float*  out0   = (float*) alloc((size_t)BB * NN * 64 * 4);
    float*  out1   = (float*) alloc((size_t)BB * NN * 128 * 4);
    float*  out2   = (float*) alloc((size_t)BB * NN * 256 * 4);
    __bf16* xcat   = (__bf16*)alloc((size_t)BB * NN * 448 * 2);
    float*  fused  = (float*) alloc((size_t)BB * NN * 384 * 4);
    float*  pooled = (float*) alloc((size_t)BB * 384 * 4);
    __bf16* pooledb= (__bf16*)alloc((size_t)BB * 384 * 2);
    float*  hbuf   = (float*) alloc((size_t)BB * 256 * 4);
    float*  counts = (float*) alloc((size_t)BB * 4);
    __bf16* w1_0 = (__bf16*)alloc(64  * 32  * 2);
    __bf16* w2_0 = (__bf16*)alloc(64  * 64  * 2);
    __bf16* w3_0 = (__bf16*)alloc(64  * 64  * 2);
    __bf16* w1_1 = (__bf16*)alloc(128 * 128 * 2);
    __bf16* w2_1 = (__bf16*)alloc(128 * 128 * 2);
    __bf16* w3_1 = (__bf16*)alloc(128 * 128 * 2);
    __bf16* w1_2 = (__bf16*)alloc(256 * 256 * 2);
    __bf16* w2_2 = (__bf16*)alloc(256 * 256 * 2);
    __bf16* w3_2 = (__bf16*)alloc(256 * 256 * 2);
    __bf16* wfu  = (__bf16*)alloc(384 * 448 * 2);
    __bf16* wfc1 = (__bf16*)alloc(256 * 384 * 2);

    // ---- raise dynamic LDS limits ----
    (void)hipFuncSetAttribute((const void*)knn_kernel,
        hipFuncAttributeMaxDynamicSharedMemorySize, (NN * 128 + NN * NN) * 4);
    (void)hipFuncSetAttribute((const void*)&edgeconv_kernel<7, 32, 64>,
        hipFuncAttributeMaxDynamicSharedMemorySize,
        128 * 32 * 2 + 2 * 128 * 64 * 2 + 8 * 7 * 4);
    (void)hipFuncSetAttribute((const void*)&edgeconv_kernel<64, 128, 128>,
        hipFuncAttributeMaxDynamicSharedMemorySize,
        128 * 128 * 2 + 2 * 128 * 128 * 2 + 8 * 64 * 4);
    (void)hipFuncSetAttribute((const void*)&edgeconv_kernel<128, 256, 256>,
        hipFuncAttributeMaxDynamicSharedMemorySize,
        128 * 256 * 2 + 2 * 128 * 256 * 2 + 8 * 128 * 4);
    (void)hipFuncSetAttribute((const void*)&gemm_kernel<448, 384>,
        hipFuncAttributeMaxDynamicSharedMemorySize, 128 * 448 * 2);
    (void)hipFuncSetAttribute((const void*)&gemm_kernel<384, 256>,
        hipFuncAttributeMaxDynamicSharedMemorySize, 128 * 384 * 2);

    // ---- 1. preprocessing ----
    prep_kernel<<<BB, NN, 0, stream>>>(points, features, mask, bn_scale, bn_bias,
                                       pts2, fts0, counts);

    // ---- 2. weight conversion ----
    auto wconv = [&](int src, __bf16* dst, int Co, int Ci, int Cip) {
        int total = Co * Cip;
        wconv_kernel<<<(total + 255) / 256, 256, 0, stream>>>(
            (const float*)d_in[src], dst, Co, Ci, Cip);
    };
    wconv(3,  w1_0, 64, 14, 32);
    wconv(6,  w2_0, 64, 64, 64);
    wconv(9,  w3_0, 64, 64, 64);
    wconv(15, w1_1, 128, 128, 128);
    wconv(18, w2_1, 128, 128, 128);
    wconv(21, w3_1, 128, 128, 128);
    wconv(27, w1_2, 256, 256, 256);
    wconv(30, w2_2, 256, 256, 256);
    wconv(33, w3_2, 256, 256, 256);
    wconv(45, wfu,  384, 448, 448);
    wconv(41, wfc1, 256, 384, 384);

    // ---- 3. block 0 : knn on 2-D coords, EdgeConv 7 -> 64 ----
    knn_kernel<<<BB, 256, (NN * 2 + NN * NN) * 4, stream>>>(pts2, mask, 2, idx);
    edgeconv_kernel<7, 32, 64><<<BB * 16, 256,
        128 * 32 * 2 + 2 * 128 * 64 * 2 + 8 * 7 * 4, stream>>>(
        fts0, idx, mask, w1_0, w2_0, w3_0,
        (const float*)d_in[5],  (const float*)d_in[4],
        (const float*)d_in[8],  (const float*)d_in[7],
        (const float*)d_in[11], (const float*)d_in[10],
        (const float*)d_in[12], (const float*)d_in[14], (const float*)d_in[13],
        out0);

    // ---- 4. block 1 : knn on 64-D feats, EdgeConv 64 -> 128 ----
    knn_kernel<<<BB, 256, (NN * 64 + NN * NN) * 4, stream>>>(out0, mask, 64, idx);
    edgeconv_kernel<64, 128, 128><<<BB * 16, 256,
        128 * 128 * 2 + 2 * 128 * 128 * 2 + 8 * 64 * 4, stream>>>(
        out0, idx, mask, w1_1, w2_1, w3_1,
        (const float*)d_in[17], (const float*)d_in[16],
        (const float*)d_in[20], (const float*)d_in[19],
        (const float*)d_in[23], (const float*)d_in[22],
        (const float*)d_in[24], (const float*)d_in[26], (const float*)d_in[25],
        out1);

    // ---- 5. block 2 : knn on 128-D feats, EdgeConv 128 -> 256 ----
    knn_kernel<<<BB, 256, (NN * 128 + NN * NN) * 4, stream>>>(out1, mask, 128, idx);
    edgeconv_kernel<128, 256, 256><<<BB * 16, 256,
        128 * 256 * 2 + 2 * 128 * 256 * 2 + 8 * 128 * 4, stream>>>(
        out1, idx, mask, w1_2, w2_2, w3_2,
        (const float*)d_in[29], (const float*)d_in[28],
        (const float*)d_in[32], (const float*)d_in[31],
        (const float*)d_in[35], (const float*)d_in[34],
        (const float*)d_in[36], (const float*)d_in[38], (const float*)d_in[37],
        out2);

    // ---- 6. concat + fusion GEMM (448 -> 384), masked ----
    {
        size_t total = (size_t)BB * NN * 448;
        concat_kernel<<<(int)((total + 255) / 256), 256, 0, stream>>>(out0, out1, out2, xcat);
    }
    gemm_kernel<448, 384><<<BB * NN / 128, 256, 128 * 448 * 2, stream>>>(
        xcat, wfu,
        (const float*)d_in[47], (const float*)d_in[46],
        /*relu=*/1, /*rowscale=*/mask, fused);

    // ---- 7. pool + fc1 (WMMA) + fc2 ----
    pool_kernel<<<BB, 256, 0, stream>>>(fused, counts, pooled, pooledb);
    gemm_kernel<384, 256><<<BB / 128, 256, 128 * 384 * 2, stream>>>(
        pooledb, wfc1,
        /*scale=*/nullptr, (const float*)d_in[42],
        /*relu=*/1, /*rowscale=*/nullptr, hbuf);
    fc2_kernel<<<(BB * 5 + 255) / 256, 256, 0, stream>>>(
        hbuf, (const float*)d_in[43], (const float*)d_in[44], (float*)d_out);
}
